// EventPCDAlignment_49555332662164
// MI455X (gfx1250) — matrix operand order repaired and checked by reading, not compiled, not run on hardware
//
#include <hip/hip_runtime.h>
#include <hip/hip_bf16.h>

// EventPCDAlignment for gfx1250 (MI455X).
// Dense convs -> implicit GEMM on v_wmma_f32_16x16x32_f16 (wave32, f32 accum),
// operands pre-packed to f16 (weights [Cout_pad][tap][Cin], activations NHWC).
// Per-block weight panel staged in LDS (ds_load_b128) to kill the 4x redundant
// A-fragment traffic; B fragments are contiguous global_load_b128s.
// Deformable conv / LSTM gates / upsample -> VALU kernels.
//
// Input order (JAX pytree = sorted dict keys):
//  0 evs1 (4,5,64,128,128)  1 evs2 (.. 64,64)  2 evs3 (.. 32,32)
//  3 img1 (4,64,128,128)    4 img2             5 img3
//  6..17  l1: dcn_b,dcn_w,fc_b,fc_w,lstm_b,lstm_w,oc2_b,oc2_w,oc3_b,oc3_w,offm_b,offm_w
//  18..29 l2: same as l1
//  30..37 l3: dcn_b,dcn_w,lstm_b,lstm_w,oc2_b,oc2_w,offm_b,offm_w
// Output: aligned1 | aligned2 | aligned3 concatenated (f32).

typedef __attribute__((ext_vector_type(16))) _Float16 v16h;
typedef __attribute__((ext_vector_type(8)))  _Float16 v8h;
typedef __attribute__((ext_vector_type(8)))  float    v8f;

#define NF 64   // num_feat C
#define NG 8    // deformable groups
#define DK 5    // dcn kernel
#define NB 4    // batch
#define NT 5    // timesteps

__device__ __forceinline__ float lrelu_f(float v) { return v >= 0.f ? v : 0.1f * v; }

// ---------------------------------------------------------------------------
// Implicit-GEMM conv2d (stride 1, zero pad KS/2) on WMMA f16->f32.
// Wave tile: 16 output channels x 16 consecutive pixels; 4 waves/block share
// one 16-row weight panel staged in LDS.
//   in : f16 NHWC (B,H,W,CIN)
//   w  : f16 packed [CoutPad][KS*KS][CIN], rows >= Cout zero-filled
//   out: f32 NCHW
// A frag (16x32): lane m-row = lane&15; elems 0-7 -> K = 8*hi + e,
//                 elems 8-15 -> K = 16 + 8*hi + (e-8)   (hi = lane>>4)
// B frag (32x16): lane col = lane&15; same K mapping as A
// Both are contiguous f16 runs in the packed layouts -> b128 accesses.
// ---------------------------------------------------------------------------
template <int KS, int CIN>
__global__ void k_conv_wmma(const _Float16* __restrict__ in, const _Float16* __restrict__ w,
                            const float* __restrict__ bias, float* __restrict__ out,
                            int B, int H, int W, int Cout, int act)
{
    constexpr int KK    = KS * KS;
    constexpr int PAD   = KS / 2;
    constexpr int WELEM = 16 * KK * CIN;     // f16 elements in the block's A panel

    __shared__ __align__(16) _Float16 wlds[WELEM];

    const int tid   = threadIdx.x;
    const int lane  = tid & 31;
    const int wave  = tid >> 5;
    const int ptile = blockIdx.x * 4 + wave;   // 4 waves / block, 16 px each
    const int co0   = blockIdx.y * 16;

    // cooperative LDS stage of the weight panel (rows co0..co0+15 are contiguous)
    {
        const _Float16* wsrc = w + (size_t)co0 * KK * CIN;
        for (int i = tid; i < WELEM / 8; i += 128)
            *(v8h*)(wlds + i * 8) = *(const v8h*)(wsrc + (size_t)i * 8);
    }
    __syncthreads();

    const int n = lane & 15;
    const int p = ptile * 16 + n;              // linear pixel (W%16==0: no row straddle)
    const int x = p % W;
    const int yb = p / W;
    const int y  = yb % H;
    const int b  = yb / H;

    const int hi     = lane >> 4;
    const int mrow   = lane & 15;
    const int khalfA = hi * 8;
    const int kbaseB = hi * 8;

    const _Float16* wrow = wlds + (mrow * KK) * CIN + khalfA;

    __builtin_prefetch(in + ((size_t)(b * H + y) * W + x) * CIN, 0, 3);

    v8f acc = {};

    for (int tap = 0; tap < KK; ++tap) {
        const int sy = y + tap / KS - PAD;
        const int sx = x + tap % KS - PAD;
        const bool inb = (sy >= 0) && (sy < H) && (sx >= 0) && (sx < W);
        const _Float16* bp = in + (((size_t)(b * H + sy)) * W + sx) * CIN + kbaseB;
        const _Float16* ap = wrow + tap * CIN;

#pragma unroll
        for (int ci0 = 0; ci0 < CIN; ci0 += 32) {
            const v8h alo = *(const v8h*)(ap + ci0);        // ds_load_b128
            const v8h ahi = *(const v8h*)(ap + ci0 + 16);
            const v16h afrag = __builtin_shufflevector(alo, ahi,
                0, 1, 2, 3, 4, 5, 6, 7, 8, 9, 10, 11, 12, 13, 14, 15);
            v16h bfrag = {};
            if (inb) {
                const v8h blo = *(const v8h*)(bp + ci0);    // global_load_b128
                const v8h bhi = *(const v8h*)(bp + ci0 + 16);
                bfrag = __builtin_shufflevector(blo, bhi,
                    0, 1, 2, 3, 4, 5, 6, 7, 8, 9, 10, 11, 12, 13, 14, 15);
            }
            acc = __builtin_amdgcn_wmma_f32_16x16x32_f16(
                false, afrag, false, bfrag, (short)0, acc, false, false);
        }
    }

    const int mbase = co0 + hi * 8;            // C/D: VGPR r holds M = r + 8*hi
#pragma unroll
    for (int r = 0; r < 8; ++r) {
        const int m = mbase + r;
        if (m < Cout) {
            float v = acc[r] + bias[m];
            if (act == 1) v = lrelu_f(v);
            out[((size_t)(b * Cout + m) * H + y) * W + x] = v;
        }
    }
}

// ---------------------------------------------------------------------------
// Pack conv weights: f32 [Cout][Cin][KK] -> f16 [CoutPad][KK][Cin], pad rows 0.
// ---------------------------------------------------------------------------
__global__ void k_pack_w(const float* __restrict__ w, _Float16* __restrict__ wh,
                         int Cout, int CoutPad, int Cin, int KK)
{
    const long idx = (long)blockIdx.x * blockDim.x + threadIdx.x;
    const long total = (long)CoutPad * KK * Cin;
    if (idx >= total) return;
    const int ci  = (int)(idx % Cin);
    const int tap = (int)((idx / Cin) % KK);
    const int co  = (int)(idx / ((long)Cin * KK));
    const float v = (co < Cout) ? w[((size_t)co * Cin + ci) * KK + tap] : 0.f;
    wh[idx] = (_Float16)v;
}

// ---------------------------------------------------------------------------
// Channel-concat two f32 NCHW sources into f16 NHWC: dst(B,HW,C0+C1).
// s0 has a custom batch stride (lets us slice evs[:,t] without a copy).
// With C1==0 this is a plain NCHW->NHWC f16 pack of s0.
// ---------------------------------------------------------------------------
__global__ void k_concat_nhwc(_Float16* __restrict__ dst, const float* __restrict__ s0,
                              long bstride0, int C0, const float* __restrict__ s1,
                              int C1, int B, int HW)
{
    const int CT = C0 + C1;
    const long idx = (long)blockIdx.x * blockDim.x + threadIdx.x;
    const long total = (long)B * HW * CT;
    if (idx >= total) return;
    const int ch = (int)(idx % CT);
    const int p  = (int)((idx / CT) % HW);
    const int b  = (int)(idx / ((long)CT * HW));
    float v;
    if (ch < C0) v = s0[(long)b * bstride0 + (size_t)ch * HW + p];
    else         v = s1[((size_t)b * C1 + (ch - C0)) * HW + p];
    dst[idx] = (_Float16)v;
}

__global__ void k_zero(float* __restrict__ p, long n)
{
    const long i = (long)blockIdx.x * blockDim.x + threadIdx.x;
    if (i < n) p[i] = 0.f;
}

// ConvLSTM gate update: z(B,4C,HW) f32 split i,f,o,g; updates c and h in place.
__global__ void k_lstm_update(const float* __restrict__ z, float* __restrict__ c,
                              float* __restrict__ h, int B, int C, int HW)
{
    const long idx = (long)blockIdx.x * blockDim.x + threadIdx.x;
    const long total = (long)B * C * HW;
    if (idx >= total) return;
    const int p  = (int)(idx % HW);
    const int ch = (int)((idx / HW) % C);
    const int b  = (int)(idx / ((long)HW * C));
    const size_t zb = (size_t)b * 4 * C * HW;
    const float iv = z[zb + (size_t)ch * HW + p];
    const float fv = z[zb + (size_t)(C + ch) * HW + p];
    const float ov = z[zb + (size_t)(2 * C + ch) * HW + p];
    const float gv = z[zb + (size_t)(3 * C + ch) * HW + p];
    const float si = 1.f / (1.f + __expf(-iv));
    const float sf = 1.f / (1.f + __expf(-fv));
    const float so = 1.f / (1.f + __expf(-ov));
    const float tg = tanhf(gv);
    const float c2 = sf * c[idx] + si * tg;
    c[idx] = c2;
    h[idx] = so * tanhf(c2);
}

// ---------------------------------------------------------------------------
// Modulated deformable conv v2 (5x5, pad 2, DG=8 groups, Cg=Og=8).
// One thread per (group, batch, pixel): 25 taps, 4-corner bilinear gather of
// 8 channels, fused sigmoid(mask) and optional LeakyReLU.
// om channels: [0,200)=dy, [200,400)=dx, [400,600)=mask; within each section
// group g / tap t lives at channel g*25+t.
// ---------------------------------------------------------------------------
__global__ void k_dcn(const float* __restrict__ x, const float* __restrict__ om,
                      const float* __restrict__ w, const float* __restrict__ bias,
                      float* __restrict__ out, int B, int H, int W, int act)
{
    const int HW = H * W;
    const long idx = (long)blockIdx.x * blockDim.x + threadIdx.x;
    const long total = (long)B * NG * HW;
    if (idx >= total) return;
    const int p = (int)(idx % HW);
    const int b = (int)((idx / HW) % B);
    const int g = (int)(idx / ((long)HW * B));
    const int xp = p % W, yp = p / W;

    float acc[8];
#pragma unroll
    for (int o = 0; o < 8; ++o) acc[o] = bias[g * 8 + o];

    const float* omb = om + (size_t)b * (3 * NG * DK * DK) * HW + p;
    const float* xg  = x + ((size_t)b * NF + g * 8) * HW;
    const int KK = DK * DK;

    for (int t = 0; t < KK; ++t) {
        const float dy = omb[(size_t)(g * KK + t) * HW];
        const float dx = omb[(size_t)(NG * KK + g * KK + t) * HW];
        const float mr = omb[(size_t)(2 * NG * KK + g * KK + t) * HW];
        const float mm = 1.f / (1.f + __expf(-mr));
        const float sy = (float)yp + (float)(t / DK - 2) + dy;
        const float sx = (float)xp + (float)(t % DK - 2) + dx;
        const float y0f = floorf(sy), x0f = floorf(sx);
        const float wy = sy - y0f, wx = sx - x0f;
        const int y0 = (int)y0f, x0 = (int)x0f;

        float s[8] = {0.f, 0.f, 0.f, 0.f, 0.f, 0.f, 0.f, 0.f};
#pragma unroll
        for (int cyi = 0; cyi < 2; ++cyi) {
#pragma unroll
            for (int cxi = 0; cxi < 2; ++cxi) {
                const int yc = y0 + cyi, xc = x0 + cxi;
                if (yc >= 0 && yc < H && xc >= 0 && xc < W) {
                    const float wgt = (cyi ? wy : 1.f - wy) * (cxi ? wx : 1.f - wx);
                    const float* xc0 = xg + (size_t)yc * W + xc;
#pragma unroll
                    for (int c = 0; c < 8; ++c) s[c] += wgt * xc0[(size_t)c * HW];
                }
            }
        }
#pragma unroll
        for (int o = 0; o < 8; ++o) {
            float ws = 0.f;
#pragma unroll
            for (int c = 0; c < 8; ++c)
                ws += s[c] * w[((size_t)(g * 8 + o) * 8 + c) * KK + t];
            acc[o] += mm * ws;
        }
    }
#pragma unroll
    for (int o = 0; o < 8; ++o) {
        float v = acc[o];
        if (act == 1) v = lrelu_f(v);
        out[((size_t)b * NF + g * 8 + o) * HW + p] = v;
    }
}

// jax.image.resize bilinear x2 (half-pixel centers; clamped-index form matches
// jax's renormalized edge weights exactly for scale 2). dst = resize(src)*scale.
__global__ void k_upsample2(const float* __restrict__ src, float* __restrict__ dst,
                            int B, int C, int H, int W, float scale)
{
    const int Ho = 2 * H, Wo = 2 * W;
    const long idx = (long)blockIdx.x * blockDim.x + threadIdx.x;
    const long total = (long)B * C * Ho * Wo;
    if (idx >= total) return;
    const int xo = (int)(idx % Wo);
    const int yo = (int)((idx / Wo) % Ho);
    const long bc = idx / ((long)Wo * Ho);
    const float sy = yo * 0.5f - 0.25f;
    const float sx = xo * 0.5f - 0.25f;
    const float y0f = floorf(sy), x0f = floorf(sx);
    const float wy = sy - y0f, wx = sx - x0f;
    const int y0 = (int)y0f, x0 = (int)x0f;
    const int y0c = min(max(y0, 0), H - 1), y1c = min(max(y0 + 1, 0), H - 1);
    const int x0c = min(max(x0, 0), W - 1), x1c = min(max(x0 + 1, 0), W - 1);
    const float* s = src + bc * (size_t)H * W;
    const float v = (1.f - wy) * ((1.f - wx) * s[(size_t)y0c * W + x0c] + wx * s[(size_t)y0c * W + x1c])
                  +        wy  * ((1.f - wx) * s[(size_t)y1c * W + x0c] + wx * s[(size_t)y1c * W + x1c]);
    dst[idx] = v * scale;
}

// ---------------------------------------------------------------------------
extern "C" void kernel_launch(void* const* d_in, const int* in_sizes, int n_in,
                              void* d_out, int out_size, void* d_ws, size_t ws_size,
                              hipStream_t stream)
{
    (void)in_sizes; (void)n_in; (void)out_size; (void)ws_size;
    const int C = NF, B = NB, T = NT;

    const float* evs[3] = {(const float*)d_in[0], (const float*)d_in[1], (const float*)d_in[2]};
    const float* img[3] = {(const float*)d_in[3], (const float*)d_in[4], (const float*)d_in[5]};

    struct P { const float *dcn_b,*dcn_w,*fc_b,*fc_w,*lstm_b,*lstm_w,
                           *oc2_b,*oc2_w,*oc3_b,*oc3_w,*offm_b,*offm_w; };
    P prm[3];
    for (int L = 0; L < 2; ++L) { // l1, l2: 12 leaves each (alphabetical)
        const int base = 6 + 12 * L;
        prm[L].dcn_b  = (const float*)d_in[base + 0];
        prm[L].dcn_w  = (const float*)d_in[base + 1];
        prm[L].fc_b   = (const float*)d_in[base + 2];
        prm[L].fc_w   = (const float*)d_in[base + 3];
        prm[L].lstm_b = (const float*)d_in[base + 4];
        prm[L].lstm_w = (const float*)d_in[base + 5];
        prm[L].oc2_b  = (const float*)d_in[base + 6];
        prm[L].oc2_w  = (const float*)d_in[base + 7];
        prm[L].oc3_b  = (const float*)d_in[base + 8];
        prm[L].oc3_w  = (const float*)d_in[base + 9];
        prm[L].offm_b = (const float*)d_in[base + 10];
        prm[L].offm_w = (const float*)d_in[base + 11];
    }
    { // l3: 8 leaves
        const int base = 30;
        prm[2].dcn_b  = (const float*)d_in[base + 0];
        prm[2].dcn_w  = (const float*)d_in[base + 1];
        prm[2].lstm_b = (const float*)d_in[base + 2];
        prm[2].lstm_w = (const float*)d_in[base + 3];
        prm[2].oc2_b  = (const float*)d_in[base + 4];
        prm[2].oc2_w  = (const float*)d_in[base + 5];
        prm[2].offm_b = (const float*)d_in[base + 6];
        prm[2].offm_w = (const float*)d_in[base + 7];
        prm[2].fc_b = prm[2].fc_w = prm[2].oc3_b = prm[2].oc3_w = nullptr;
    }

    // ---- workspace arena ----
    float* ws = (float*)d_ws;
    const size_t HW1 = 128 * 128;
    size_t o = 0;
    float* bufOm  = ws + o; o += (size_t)B * 608 * HW1;   // om (600ch) / lstm z (256ch)
    float* bufZ   = bufOm;
    float* bufH   = ws + o; o += (size_t)B * C * HW1;
    float* bufC   = ws + o; o += (size_t)B * C * HW1;
    float* bufOff = ws + o; o += (size_t)B * C * HW1;     // offset feature chain
    float* bufTmp = ws + o; o += (size_t)B * C * HW1;     // oc2 out / dcn feat
    float* bufFeat   = bufTmp;
    float* bufUpOff  = ws + o; o += (size_t)B * C * HW1;
    float* bufUpFeat = ws + o; o += (size_t)B * C * HW1;

    _Float16* h16 = (_Float16*)(ws + o);
    size_t oh = 0;
    _Float16* bufInH = h16 + oh; oh += (size_t)B * HW1 * 2 * C;     // conv input (NHWC f16)
    _Float16 *lstm_wh[3], *oc2_wh[3], *oc3_wh[2], *fc_wh[2], *offm_wh[3];
    for (int L = 0; L < 3; ++L) { lstm_wh[L] = h16 + oh; oh += (size_t)256 * 9 * 128; }
    for (int L = 0; L < 3; ++L) { oc2_wh[L]  = h16 + oh; oh += (size_t)64 * 9 * ((L < 2) ? 128 : 64); }
    for (int L = 0; L < 2; ++L) { oc3_wh[L]  = h16 + oh; oh += (size_t)64 * 9 * 64; }
    for (int L = 0; L < 2; ++L) { fc_wh[L]   = h16 + oh; oh += (size_t)64 * 9 * 128; }
    for (int L = 0; L < 3; ++L) { offm_wh[L] = h16 + oh; oh += (size_t)608 * 25 * 64; }

    float* outp[3];
    outp[0] = (float*)d_out;
    outp[1] = outp[0] + (size_t)B * C * HW1;
    outp[2] = outp[1] + (size_t)B * C * (HW1 / 4);

    auto ew = [&](long n) { return dim3((unsigned)((n + 255) / 256)); };
    auto packw = [&](const float* w, _Float16* wh, int Cout, int CoutPad, int Cin, int KK) {
        const long n = (long)CoutPad * KK * Cin;
        k_pack_w<<<ew(n), 256, 0, stream>>>(w, wh, Cout, CoutPad, Cin, KK);
    };

    // ---- pack all conv weights to f16 [CoutPad][tap][Cin] ----
    for (int L = 0; L < 3; ++L) {
        packw(prm[L].lstm_w, lstm_wh[L], 256, 256, 128, 9);
        packw(prm[L].oc2_w,  oc2_wh[L],  64, 64, (L < 2) ? 128 : 64, 9);
        packw(prm[L].offm_w, offm_wh[L], 600, 608, 64, 25);
        if (L < 2) {
            packw(prm[L].oc3_w, oc3_wh[L], 64, 64, 64, 9);
            packw(prm[L].fc_w,  fc_wh[L],  64, 64, 128, 9);
        }
    }

    auto conv = [&](const _Float16* in, const _Float16* wh, const float* bias, float* out,
                    int H, int W, int Cin, int Cout, int ks, int act) {
        dim3 grid((unsigned)((B * H * W) / 64), (unsigned)((Cout + 15) / 16));
        if (ks == 3 && Cin == 128)
            k_conv_wmma<3, 128><<<grid, 128, 0, stream>>>(in, wh, bias, out, B, H, W, Cout, act);
        else if (ks == 3 && Cin == 64)
            k_conv_wmma<3, 64><<<grid, 128, 0, stream>>>(in, wh, bias, out, B, H, W, Cout, act);
        else
            k_conv_wmma<5, 64><<<grid, 128, 0, stream>>>(in, wh, bias, out, B, H, W, Cout, act);
    };

    for (int i = 3; i >= 1; --i) {
        const int H = 128 >> (i - 1), W = H, HW = H * W;
        const P& p = prm[i - 1];
        const long nz = (long)B * C * HW;
        const long ncat = (long)B * HW * 2 * C;
        const long npk  = (long)B * HW * C;

        // --- ConvLSTM over T steps ---
        k_zero<<<ew(nz), 256, 0, stream>>>(bufH, nz);
        k_zero<<<ew(nz), 256, 0, stream>>>(bufC, nz);
        for (int t = 0; t < T; ++t) {
            k_concat_nhwc<<<ew(ncat), 256, 0, stream>>>(bufInH, evs[i - 1] + (size_t)t * C * HW,
                                                        (long)T * C * HW, C, bufH, C, B, HW);
            conv(bufInH, lstm_wh[i - 1], p.lstm_b, bufZ, H, W, 2 * C, 4 * C, 3, 0);
            k_lstm_update<<<ew(nz), 256, 0, stream>>>(bufZ, bufC, bufH, B, C, HW);
        }

        // --- offset feature chain ---
        if (i == 3) {
            k_concat_nhwc<<<ew(npk), 256, 0, stream>>>(bufInH, bufH, (long)C * HW, C,
                                                       nullptr, 0, B, HW);
            conv(bufInH, oc2_wh[2], p.oc2_b, bufOff, H, W, C, C, 3, 1);
        } else {
            k_concat_nhwc<<<ew(ncat), 256, 0, stream>>>(bufInH, bufH, (long)C * HW, C,
                                                        bufUpOff, C, B, HW);
            conv(bufInH, oc2_wh[i - 1], p.oc2_b, bufTmp, H, W, 2 * C, C, 3, 1);
            k_concat_nhwc<<<ew(npk), 256, 0, stream>>>(bufInH, bufTmp, (long)C * HW, C,
                                                       nullptr, 0, B, HW);
            conv(bufInH, oc3_wh[i - 1], p.oc3_b, bufOff, H, W, C, C, 3, 1);
        }

        // --- offset/mask conv (600ch, 5x5) + deformable conv ---
        k_concat_nhwc<<<ew(npk), 256, 0, stream>>>(bufInH, bufOff, (long)C * HW, C,
                                                   nullptr, 0, B, HW);
        conv(bufInH, offm_wh[i - 1], p.offm_b, bufOm, H, W, C, 3 * NG * DK * DK, 5, 0);

        const long nd = (long)B * NG * HW;
        if (i == 3) {
            k_dcn<<<ew(nd), 256, 0, stream>>>(img[2], bufOm, p.dcn_w, p.dcn_b,
                                              outp[2], B, H, W, 1);
        } else {
            k_dcn<<<ew(nd), 256, 0, stream>>>(img[i - 1], bufOm, p.dcn_w, p.dcn_b,
                                              bufFeat, B, H, W, 0);
            k_concat_nhwc<<<ew(ncat), 256, 0, stream>>>(bufInH, bufFeat, (long)C * HW, C,
                                                        bufUpFeat, C, B, HW);
            conv(bufInH, fc_wh[i - 1], p.fc_b, outp[i - 1], H, W, 2 * C, C, 3,
                 (i == 2) ? 1 : 0);
        }

        // --- feed next (finer) level ---
        if (i > 1) {
            const long nu = (long)B * C * HW * 4;
            k_upsample2<<<ew(nu), 256, 0, stream>>>(bufOff, bufUpOff, B, C, H, W, 2.0f);
            k_upsample2<<<ew(nu), 256, 0, stream>>>(outp[i - 1], bufUpFeat, B, C, H, W, 1.0f);
        }
    }
}
